// QKVAttention_12524124635615
// MI455X (gfx1250) — matrix operand order repaired and checked by reading, not compiled, not run on hardware
//
#include <hip/hip_runtime.h>
#include <hip/hip_bf16.h>
#include <math.h>

typedef __bf16 bf16_t;
typedef __attribute__((ext_vector_type(16))) __bf16 v16bf;
typedef __attribute__((ext_vector_type(8)))  __bf16 v8bf;
typedef __attribute__((ext_vector_type(8)))  float  v8f;

#define B_ 2
#define T_ 2048
#define C_ 1024
#define H_ 16
#define D_ 64

// ---------------------------------------------------------------------------
// WMMA helper: D = A(16x32 bf16) x B(32x16 bf16) + C(16x16 f32)
// ---------------------------------------------------------------------------
__device__ __forceinline__ v8f wmma_bf16(v16bf a, v16bf b, v8f acc) {
  return __builtin_amdgcn_wmma_f32_16x16x32_bf16(
      /*neg_a=*/false, a, /*neg_b=*/false, b,
      /*c_mod=*/(short)0, acc, /*reuse_a=*/false, /*reuse_b=*/false);
}

// A fragment (16x32, MxK) from row-major [lda] bf16 matrix, tile origin (row0, k0).
// Lane (m = lane&15, hi = lane>>4): element e -> K = k0 + 16*(e>>3) + 8*hi + (e&7).
__device__ __forceinline__ v16bf load_a(const bf16_t* __restrict__ Ab, int lda,
                                        int row0, int k0, int m, int hi) {
  const bf16_t* p = Ab + (size_t)(row0 + m) * lda + k0;
  v8bf lo = *(const v8bf*)(p + 8 * hi);
  v8bf hh = *(const v8bf*)(p + 16 + 8 * hi);
  v16bf a;
#pragma unroll
  for (int i = 0; i < 8; ++i) { a[i] = lo[i]; a[8 + i] = hh[i]; }
  return a;
}

// B fragment (32x16, KxN) where B[k][n] = M[n0+n][k] (K-contiguous storage).
// Lane (n = lane&15, hi = lane>>4): element e -> K = k0 + 16*(e>>3) + 8*hi + (e&7).
__device__ __forceinline__ v16bf load_bt(const bf16_t* __restrict__ Mb, int ldm,
                                         int n0, int k0, int nl, int hi) {
  const bf16_t* p = Mb + (size_t)(n0 + nl) * ldm + k0;
  v8bf lo = *(const v8bf*)(p + 8 * hi);
  v8bf hh = *(const v8bf*)(p + 16 + 8 * hi);
  v16bf b;
#pragma unroll
  for (int i = 0; i < 8; ++i) { b[i] = lo[i]; b[8 + i] = hh[i]; }
  return b;
}

// ---------------------------------------------------------------------------
// f32 -> bf16 conversion kernels
// ---------------------------------------------------------------------------
__global__ void k_cvt(const float* __restrict__ in, bf16_t* __restrict__ out, int n) {
  int i = blockIdx.x * blockDim.x + threadIdx.x;
  if (i < n) out[i] = (bf16_t)in[i];
}

// w_o [HV=1024][C=1024] -> wot [C][HV] (bf16), so the oproj B-side is K-contiguous.
__global__ void k_cvt_wo_t(const float* __restrict__ wo, bf16_t* __restrict__ wot) {
  int i = blockIdx.x * blockDim.x + threadIdx.x;
  if (i < H_ * D_ * C_) {
    int hv = i >> 10, c = i & (C_ - 1);
    wot[(size_t)c * (H_ * D_) + hv] = (bf16_t)wo[i];
  }
}

// ---------------------------------------------------------------------------
// QKV projection, 32x32 register block per wave (2 A-frags x 2 B-frags -> 4 WMMA
// per K-step; 2 b128 loads per WMMA instead of 4).
// Q/K[t][d] = sum_c x[t][c] * w[d][c];  V stored transposed [d][t].
// Grid: (T/32, D/32, 3*B*H), block = 1 wave.
// ---------------------------------------------------------------------------
__global__ void k_qkv(const bf16_t* __restrict__ xb,
                      const bf16_t* __restrict__ wqb,
                      const bf16_t* __restrict__ wkb,
                      const bf16_t* __restrict__ wvb,
                      bf16_t* __restrict__ Qb, bf16_t* __restrict__ Kb,
                      bf16_t* __restrict__ Vt) {
  const int lane = threadIdx.x;
  const int m = lane & 15, hi = lane >> 4;
  const int t0 = blockIdx.x * 32;
  const int d0 = blockIdx.y * 32;
  const int z = blockIdx.z;                 // [0, 3*B*H)
  const int mat = z / (B_ * H_);
  const int nh  = z % (B_ * H_);
  const int n = nh / H_, h = nh % H_;

  const bf16_t* X = xb + (size_t)n * T_ * C_;
  const bf16_t* W = (mat == 0 ? wqb : (mat == 1 ? wkb : wvb)) + (size_t)h * D_ * C_;

  v8f acc00 = {}, acc01 = {}, acc10 = {}, acc11 = {};
  for (int kk = 0; kk < C_; kk += 32) {
    v16bf a0 = load_a(X, C_, t0,      kk, m, hi);
    v16bf a1 = load_a(X, C_, t0 + 16, kk, m, hi);
    v16bf b0 = load_bt(W, C_, d0,      kk, m, hi);   // B[k=c][n=d] = W[d][c]
    v16bf b1 = load_bt(W, C_, d0 + 16, kk, m, hi);
    acc00 = wmma_bf16(a0, b0, acc00);
    acc01 = wmma_bf16(a0, b1, acc01);
    acc10 = wmma_bf16(a1, b0, acc10);
    acc11 = wmma_bf16(a1, b1, acc11);
  }

  if (mat == 2) {
    // Vt[nh][d][t]; lane holds column d, rows t = .. + 8*hi + r (contiguous).
    bf16_t* Vh = Vt + (size_t)nh * D_ * T_;
    v8f* accs[4] = {&acc00, &acc01, &acc10, &acc11};
#pragma unroll
    for (int ti = 0; ti < 2; ++ti)
#pragma unroll
      for (int dj = 0; dj < 2; ++dj) {
        v8f a = *accs[ti * 2 + dj];
        v8bf o;
#pragma unroll
        for (int r = 0; r < 8; ++r) o[r] = (bf16_t)a[r];
        *(v8bf*)(Vh + (size_t)(d0 + 16 * dj + m) * T_ + t0 + 16 * ti + 8 * hi) = o;
      }
  } else {
    bf16_t* O = (mat == 0 ? Qb : Kb) + (size_t)nh * T_ * D_;
    const float s = (mat == 0) ? 0.125f : 1.0f;      // fold 1/sqrt(DQK) into Q (exact)
    v8f* accs[4] = {&acc00, &acc01, &acc10, &acc11};
#pragma unroll
    for (int ti = 0; ti < 2; ++ti)
#pragma unroll
      for (int dj = 0; dj < 2; ++dj) {
        v8f a = *accs[ti * 2 + dj];
#pragma unroll
        for (int r = 0; r < 8; ++r)
          O[(size_t)(t0 + 16 * ti + r + 8 * hi) * D_ + d0 + 16 * dj + m] =
              (bf16_t)(a[r] * s);
      }
  }
}

// ---------------------------------------------------------------------------
// Causal flash attention: one wave per (n,h, 16-row q-tile); online softmax over
// 32-key blocks. Q-frags stay resident, so reuse is already 8 WMMA / 16 loads.
// Grid: (T/16, B*H), block = 1 wave.
// ---------------------------------------------------------------------------
__global__ void k_flash(const bf16_t* __restrict__ Qb,
                        const bf16_t* __restrict__ Kb,
                        const bf16_t* __restrict__ Vt,
                        bf16_t* __restrict__ Yb) {
  __shared__ bf16_t plds[16 * 32];                   // P tile bounce (C/D -> A layout)
  const int lane = threadIdx.x;
  const int m = lane & 15, hi = lane >> 4;
  const int qi = blockIdx.x;                         // 0..127
  const int nh = blockIdx.y;                         // 0..31
  const int n = nh / H_, h = nh % H_;
  const int t0 = qi * 16;

  const bf16_t* Q = Qb + (size_t)nh * T_ * D_;
  const bf16_t* K = Kb + (size_t)nh * T_ * D_;
  const bf16_t* V = Vt + (size_t)nh * D_ * T_;

  // Q fragments (rows t0..t0+15, K = 64 -> two 32-wide chunks), resident in VGPRs.
  v16bf qa0 = load_a(Q, D_, t0, 0, m, hi);
  v16bf qa1 = load_a(Q, D_, t0, 32, m, hi);

  v8f acc0 = {}, acc1 = {}, acc2 = {}, acc3 = {};
  float mrow[8], lrow[8];
#pragma unroll
  for (int r = 0; r < 8; ++r) { mrow[r] = -INFINITY; lrow[r] = 0.0f; }

  const int njb = qi / 2 + 1;                        // 32-key blocks covering s <= t
  for (int jb = 0; jb < njb; ++jb) {
    const int s0 = jb * 32;
    // S0 = Q * K[s0..s0+15]^T,  S1 = Q * K[s0+16..s0+31]^T  (B[k=d][n=s] = K[s][d])
    v8f sa = {}, sb = {};
    sa = wmma_bf16(qa0, load_bt(K, D_, s0,      0,  m, hi), sa);
    sa = wmma_bf16(qa1, load_bt(K, D_, s0,      32, m, hi), sa);
    sb = wmma_bf16(qa0, load_bt(K, D_, s0 + 16, 0,  m, hi), sb);
    sb = wmma_bf16(qa1, load_bt(K, D_, s0 + 16, 32, m, hi), sb);

    // Causal mask + online softmax. C/D layout: lane = col (key), VGPR r = row t.
    float p0[8], p1[8], alpha[8];
#pragma unroll
    for (int r = 0; r < 8; ++r) {
      const int t = t0 + r + 8 * hi;
      float v0 = (s0 + m      > t) ? -INFINITY : sa[r];
      float v1 = (s0 + 16 + m > t) ? -INFINITY : sb[r];
      float mx = fmaxf(v0, v1);
#pragma unroll
      for (int msk = 1; msk < 16; msk <<= 1) mx = fmaxf(mx, __shfl_xor(mx, msk, 32));
      float mnew = fmaxf(mrow[r], mx);
      alpha[r] = __expf(mrow[r] - mnew);
      p0[r] = __expf(v0 - mnew);
      p1[r] = __expf(v1 - mnew);
      float sum = p0[r] + p1[r];
#pragma unroll
      for (int msk = 1; msk < 16; msk <<= 1) sum += __shfl_xor(sum, msk, 32);
      lrow[r] = lrow[r] * alpha[r] + sum;
      mrow[r] = mnew;
    }
    // Rescale running output by alpha (per-row, no cross-lane traffic needed).
#pragma unroll
    for (int r = 0; r < 8; ++r) {
      acc0[r] *= alpha[r]; acc1[r] *= alpha[r];
      acc2[r] *= alpha[r]; acc3[r] *= alpha[r];
    }
    // P (f32, C/D layout) -> LDS row-major bf16 [16][32] -> A fragment.
    // DS ops are in-order within a wave, so no barrier is needed (blockDim = 32).
#pragma unroll
    for (int r = 0; r < 8; ++r) {
      plds[(r + 8 * hi) * 32 + m]      = (bf16_t)p0[r];
      plds[(r + 8 * hi) * 32 + 16 + m] = (bf16_t)p1[r];
    }
    v8bf plo = *(const v8bf*)(plds + m * 32 + 8 * hi);
    v8bf phi = *(const v8bf*)(plds + m * 32 + 16 + 8 * hi);
    v16bf pa;
#pragma unroll
    for (int i = 0; i < 8; ++i) { pa[i] = plo[i]; pa[8 + i] = phi[i]; }

    // O += P(16x32) * V(32x64); B[k=s][n=dv] = Vt[dv][s0+k] (K-contiguous).
    acc0 = wmma_bf16(pa, load_bt(V, T_, 0,  s0, m, hi), acc0);
    acc1 = wmma_bf16(pa, load_bt(V, T_, 16, s0, m, hi), acc1);
    acc2 = wmma_bf16(pa, load_bt(V, T_, 32, s0, m, hi), acc2);
    acc3 = wmma_bf16(pa, load_bt(V, T_, 48, s0, m, hi), acc3);
  }

  // Normalize and store Y[n][t][h*64 + dv] (bf16 feed for the output projection).
  float inv[8];
#pragma unroll
  for (int r = 0; r < 8; ++r) inv[r] = 1.0f / lrow[r];
  bf16_t* Y = Yb + ((size_t)n * T_ + t0) * (H_ * D_) + h * D_;
#pragma unroll
  for (int r = 0; r < 8; ++r) {
    size_t row = (size_t)(r + 8 * hi) * (H_ * D_);
    Y[row + 0  + m] = (bf16_t)(acc0[r] * inv[r]);
    Y[row + 16 + m] = (bf16_t)(acc1[r] * inv[r]);
    Y[row + 32 + m] = (bf16_t)(acc2[r] * inv[r]);
    Y[row + 48 + m] = (bf16_t)(acc3[r] * inv[r]);
  }
}

// ---------------------------------------------------------------------------
// Output projection, 32x32 register block per wave:
// out[nt][c] = sum_hv Y[nt][hv] * wo[hv][c], f32 output.
// Grid: (B*T/32, C/32), block = 1 wave.
// ---------------------------------------------------------------------------
__global__ void k_oproj(const bf16_t* __restrict__ Yb,
                        const bf16_t* __restrict__ wot,
                        float* __restrict__ out) {
  const int lane = threadIdx.x;
  const int m = lane & 15, hi = lane >> 4;
  const int r0 = blockIdx.x * 32;                    // rows over B*T
  const int c0 = blockIdx.y * 32;                    // output channels
  v8f acc00 = {}, acc01 = {}, acc10 = {}, acc11 = {};
  for (int kk = 0; kk < H_ * D_; kk += 32) {
    v16bf a0 = load_a(Yb, H_ * D_, r0,      kk, m, hi);
    v16bf a1 = load_a(Yb, H_ * D_, r0 + 16, kk, m, hi);
    v16bf b0 = load_bt(wot, H_ * D_, c0,      kk, m, hi); // B[k=hv][n=c] = wot[c][hv]
    v16bf b1 = load_bt(wot, H_ * D_, c0 + 16, kk, m, hi);
    acc00 = wmma_bf16(a0, b0, acc00);
    acc01 = wmma_bf16(a0, b1, acc01);
    acc10 = wmma_bf16(a1, b0, acc10);
    acc11 = wmma_bf16(a1, b1, acc11);
  }
  v8f* accs[4] = {&acc00, &acc01, &acc10, &acc11};
#pragma unroll
  for (int ti = 0; ti < 2; ++ti)
#pragma unroll
    for (int cj = 0; cj < 2; ++cj) {
      v8f a = *accs[ti * 2 + cj];
#pragma unroll
      for (int r = 0; r < 8; ++r)
        out[(size_t)(r0 + 16 * ti + r + 8 * hi) * C_ + c0 + 16 * cj + m] = a[r];
    }
}

// ---------------------------------------------------------------------------
extern "C" void kernel_launch(void* const* d_in, const int* in_sizes, int n_in,
                              void* d_out, int out_size, void* d_ws, size_t ws_size,
                              hipStream_t stream) {
  const float* x  = (const float*)d_in[0];
  const float* wq = (const float*)d_in[1];
  const float* wk = (const float*)d_in[2];
  const float* wv = (const float*)d_in[3];
  const float* wo = (const float*)d_in[4];
  float* out = (float*)d_out;

  // Workspace layout (bf16), 48 MB total.
  char* ws = (char*)d_ws;
  bf16_t* xb  = (bf16_t*)(ws);                       //  8 MB  x
  bf16_t* wqb = (bf16_t*)(ws + ( 8u << 20));         //  2 MB
  bf16_t* wkb = (bf16_t*)(ws + (10u << 20));         //  2 MB
  bf16_t* wvb = (bf16_t*)(ws + (12u << 20));         //  2 MB
  bf16_t* wot = (bf16_t*)(ws + (14u << 20));         //  2 MB  w_o transposed
  bf16_t* Qb  = (bf16_t*)(ws + (16u << 20));         //  8 MB  [B,H,T,D] (pre-scaled)
  bf16_t* Kb  = (bf16_t*)(ws + (24u << 20));         //  8 MB  [B,H,T,D]
  bf16_t* Vt  = (bf16_t*)(ws + (32u << 20));         //  8 MB  [B,H,D,T] (transposed)
  bf16_t* Yb  = (bf16_t*)(ws + (40u << 20));         //  8 MB  [B,T,H*D]

  const int NX = B_ * T_ * C_;                       // 4194304
  const int NW = H_ * D_ * C_;                       // 1048576
  k_cvt<<<dim3((NX + 255) / 256), dim3(256), 0, stream>>>(x,  xb,  NX);
  k_cvt<<<dim3((NW + 255) / 256), dim3(256), 0, stream>>>(wq, wqb, NW);
  k_cvt<<<dim3((NW + 255) / 256), dim3(256), 0, stream>>>(wk, wkb, NW);
  k_cvt<<<dim3((NW + 255) / 256), dim3(256), 0, stream>>>(wv, wvb, NW);
  k_cvt_wo_t<<<dim3((NW + 255) / 256), dim3(256), 0, stream>>>(wo, wot);

  k_qkv<<<dim3(T_ / 32, D_ / 32, 3 * B_ * H_), dim3(32), 0, stream>>>(
      xb, wqb, wkb, wvb, Qb, Kb, Vt);
  k_flash<<<dim3(T_ / 16, B_ * H_), dim3(32), 0, stream>>>(Qb, Kb, Vt, Yb);
  k_oproj<<<dim3(B_ * T_ / 32, C_ / 32), dim3(32), 0, stream>>>(Yb, wot, out);
}
